// RNNModel_37288906064334
// MI455X (gfx1250) — matrix operand order repaired and checked by reading, not compiled, not run on hardware
//
#include <hip/hip_runtime.h>

typedef __attribute__((ext_vector_type(16))) _Float16 v16h;
typedef __attribute__((ext_vector_type(8)))  _Float16 v8h;
typedef __attribute__((ext_vector_type(8)))  float    v8f;

#define SEQ    8192
#define EMBED  27
#define HIDDEN 30

__device__ __forceinline__ float fast_sigmoid(float x) {
    return __builtin_amdgcn_rcpf(1.0f + __expf(-x));
}
__device__ __forceinline__ float fast_tanh(float x) {
    return 1.0f - 2.0f * __builtin_amdgcn_rcpf(__expf(2.0f * x) + 1.0f);
}

// ---------------------------------------------------------------------------
// Phase 1: gxw[t][n][j] = (emb[tok[t]] . W_ih[row] + b_ih[row] + b_hh[row])
// for padded row rp = j*16+n, rp in [0,128): group p = rp/32, e = rp%32,
// row = p*30+e if e<30 else value 0.  One block per timestep.
// ---------------------------------------------------------------------------
__global__ __launch_bounds__(128) void gx_precompute(
    const int*   __restrict__ tokens,
    const float* __restrict__ emb,
    const float* __restrict__ W_ih,
    const float* __restrict__ b_ih,
    const float* __restrict__ b_hh,
    float*       __restrict__ gxw)
{
    __shared__ float x[32];
    const int t  = blockIdx.x;
    const int rp = threadIdx.x;           // padded row 0..127

    const int tok = tokens[t];
    if (rp < EMBED) x[rp] = emb[(size_t)tok * EMBED + rp];
    __syncthreads();

    const int p = rp >> 5;                // gate group (i,f,g,o)
    const int e = rp & 31;
    float acc = 0.0f;
    if (e < HIDDEN) {
        const int r = p * HIDDEN + e;     // original gate row 0..119
        acc = b_ih[r] + b_hh[r];
        #pragma unroll
        for (int k = 0; k < EMBED; ++k)
            acc += x[k] * W_ih[r * EMBED + k];
    }
    const int n = rp & 15;
    const int j = rp >> 4;
    gxw[(size_t)t * 128 + n * 8 + j] = acc;
}

// ---------------------------------------------------------------------------
// Phase 2: single-wave LSTM scan.  Weight tiles live in VGPRs as WMMA B
// operands; h is rebuilt each step into the A (16x32 f16) layout via LDS.
// gx loads are software-pipelined one step ahead; all 8 accumulators are
// forced live simultaneously so the 8 independent WMMAs issue back-to-back.
// ---------------------------------------------------------------------------
__global__ __launch_bounds__(32) void lstm_scan(
    const float* __restrict__ gxw,
    const float* __restrict__ W_hh,
    const float* __restrict__ W_lin,
    const float* __restrict__ b_lin,
    float*       __restrict__ out)
{
    // halves 0..29 : h (f16).  halves 30..63 : permanent zeros (pad + zero
    // rows for lanes != 0,16).  halves 64..127 : dump region for redirected
    // garbage stores from inactive lanes.
    __shared__ _Float16 hbuf[128];
    const int lane = threadIdx.x;

    hbuf[lane]      = (_Float16)0.0f;
    hbuf[lane + 32] = (_Float16)0.0f;
    hbuf[lane + 64] = (_Float16)0.0f;
    hbuf[lane + 96] = (_Float16)0.0f;

    const int n     = lane & 15;
    const int kbase = (lane >> 4) << 4;      // 0 or 16

    // Build the 8 resident B tiles (32x16 f16 each, 8 VGPRs/lane).
    v16h B0, B1, B2, B3, B4, B5, B6, B7;
#define LOADB(J, BV)                                                        \
    {                                                                       \
        const int rp = (J) * 16 + n;                                        \
        const int p  = rp >> 5;                                             \
        const int e  = rp & 31;                                             \
        _Pragma("unroll")                                                   \
        for (int idx = 0; idx < 16; ++idx) {                                \
            const int k = kbase + idx;                                      \
            float w = 0.0f;                                                 \
            if (e < HIDDEN && k < HIDDEN)                                   \
                w = W_hh[(p * HIDDEN + e) * HIDDEN + k];                    \
            BV[idx] = (_Float16)w;                                          \
        }                                                                   \
    }
    LOADB(0, B0) LOADB(1, B1) LOADB(2, B2) LOADB(3, B3)
    LOADB(4, B4) LOADB(5, B5) LOADB(6, B6) LOADB(7, B7)
#undef LOADB

    char* hb = (char*)hbuf;
    // A-operand load addresses (bytes): lanes other than 0/16 read zeros.
    const int a0 = (lane == 0) ? 0 : (lane == 16) ? 32 : 64;
    const int a1 = a0 + 16;
    // h store addresses; inactive lanes redirected to dump region (>=128B).
    const int s0 = (lane < 16) ? 2 * lane        : 128 + 2 * lane;
    const int s1 = (lane < 14) ? 32 + 2 * lane   : 192 + 2 * lane;

    const float* gxl = gxw + (size_t)n * 8;

    float c0 = 0.0f, c1 = 0.0f, h0v = 0.0f, h1v = 0.0f;
    const v8f cz = {};

    __syncthreads();

    // Pipeline prologue: gx for step 0.
    float4 ga = *(const float4*)(gxl);
    float4 gb = *(const float4*)(gxl + 4);

    for (int t = 0; t < SEQ; ++t) {
        // Issue next step's gx loads early (independent of the recurrence).
        const int tn = (t + 1 < SEQ) ? (t + 1) : 0;
        const float4 ga_n = *(const float4*)(gxl + (size_t)tn * 128);
        const float4 gb_n = *(const float4*)(gxl + (size_t)tn * 128 + 4);
        __builtin_prefetch(gxl + (size_t)(t + 32) * 128, 0, 1);

        // Rebuild A (h broadcast into row 0) from LDS.
        v8h alo = *(const v8h*)(hb + a0);
        v8h ahi = *(const v8h*)(hb + a1);
        v16h A  = __builtin_shufflevector(alo, ahi,
                    0, 1, 2, 3, 4, 5, 6, 7, 8, 9, 10, 11, 12, 13, 14, 15);

        // 8 independent 16x16x32 f16 WMMAs: gates_padded = h * W_hh_pad^T
        v8f d0 = __builtin_amdgcn_wmma_f32_16x16x32_f16(false, A, false, B0, (short)0, cz, false, false);
        v8f d1 = __builtin_amdgcn_wmma_f32_16x16x32_f16(false, A, false, B1, (short)0, cz, false, false);
        v8f d2 = __builtin_amdgcn_wmma_f32_16x16x32_f16(false, A, false, B2, (short)0, cz, false, false);
        v8f d3 = __builtin_amdgcn_wmma_f32_16x16x32_f16(false, A, false, B3, (short)0, cz, false, false);
        v8f d4 = __builtin_amdgcn_wmma_f32_16x16x32_f16(false, A, false, B4, (short)0, cz, false, false);
        v8f d5 = __builtin_amdgcn_wmma_f32_16x16x32_f16(false, A, false, B5, (short)0, cz, false, false);
        v8f d6 = __builtin_amdgcn_wmma_f32_16x16x32_f16(false, A, false, B6, (short)0, cz, false, false);
        v8f d7 = __builtin_amdgcn_wmma_f32_16x16x32_f16(false, A, false, B7, (short)0, cz, false, false);
        // Force all 8 accumulators live in distinct register tuples so the
        // WMMAs have no WAW hazards and issue back-to-back; pay the
        // WMMA->VALU hazard window once, after the last WMMA.
        asm volatile("" : "+v"(d0), "+v"(d1), "+v"(d2), "+v"(d3),
                          "+v"(d4), "+v"(d5), "+v"(d6), "+v"(d7));

        // Lane n owns gate elements e=n (d0,d2,d4,d6) and e=16+n (d1,d3,d5,d7).
        const float gi0 = d0[0] + ga.x;
        const float gi1 = d1[0] + ga.y;
        const float gf0 = d2[0] + ga.z;
        const float gf1 = d3[0] + ga.w;
        const float gg0 = d4[0] + gb.x;
        const float gg1 = d5[0] + gb.y;
        const float go0 = d6[0] + gb.z;
        const float go1 = d7[0] + gb.w;

        c0  = fast_sigmoid(gf0) * c0 + fast_sigmoid(gi0) * fast_tanh(gg0);
        c1  = fast_sigmoid(gf1) * c1 + fast_sigmoid(gi1) * fast_tanh(gg1);
        h0v = fast_sigmoid(go0) * fast_tanh(c0);
        h1v = fast_sigmoid(go1) * fast_tanh(c1);

        // Publish h (f16) for next step's A build; garbage lanes hit dump.
        *(_Float16*)(hb + s0) = (_Float16)h0v;
        *(_Float16*)(hb + s1) = (_Float16)h1v;
        __syncthreads();   // single-wave WG: barrier ~ NOP, keeps DS waits

        ga = ga_n;
        gb = gb_n;
    }

    // Head: out = h . W_lin + b_lin   (h[lane] in h0v, h[16+lane] in h1v)
    float part = 0.0f;
    if (lane < 16) part  = h0v * W_lin[lane];
    if (lane < 14) part += h1v * W_lin[16 + lane];
    #pragma unroll
    for (int off = 16; off > 0; off >>= 1)
        part += __shfl_down(part, off, 32);
    if (lane == 0) out[0] = part + b_lin[0];
}

// ---------------------------------------------------------------------------
extern "C" void kernel_launch(void* const* d_in, const int* in_sizes, int n_in,
                              void* d_out, int out_size, void* d_ws, size_t ws_size,
                              hipStream_t stream) {
    const int*   tokens = (const int*)  d_in[0];
    const float* emb    = (const float*)d_in[1];
    const float* W_ih   = (const float*)d_in[2];
    const float* W_hh   = (const float*)d_in[3];
    const float* b_ih   = (const float*)d_in[4];
    const float* b_hh   = (const float*)d_in[5];
    const float* W_lin  = (const float*)d_in[6];
    const float* b_lin  = (const float*)d_in[7];
    float* out = (float*)d_out;
    float* gxw = (float*)d_ws;            // SEQ*128 floats = 4 MB scratch

    gx_precompute<<<SEQ, 128, 0, stream>>>(tokens, emb, W_ih, b_ih, b_hh, gxw);
    lstm_scan<<<1, 32, 0, stream>>>(gxw, W_hh, W_lin, b_lin, out);
    (void)in_sizes; (void)n_in; (void)out_size; (void)ws_size;
}
